// _QuantumGate_65481071395894
// MI455X (gfx1250) — matrix operand order repaired and checked by reading, not compiled, not run on hardware
//
#include <hip/hip_runtime.h>
#include <math.h>

typedef __attribute__((ext_vector_type(16))) _Float16 v16h;
typedef __attribute__((ext_vector_type(8)))  float    v8f;

#define N_WIRES 10

// One wave32 handles a tile of 16 batch rows.
//   out[b,k] = prod_{j<=k} cos(x[b,j]) * cos(params[j])
// computed as sign-parity (WMMA #3) * exp(prefix-sum of log|c|) where the
// prefix sum runs through v_wmma_f32_16x16x32_f16 twice (hi/lo compensated
// split of the fp32 logs against an exact 0/1 lower-triangular mask).
__global__ __launch_bounds__(256)
void qsim_prefix_wmma(const float* __restrict__ x,
                      const float* __restrict__ params,
                      float* __restrict__ out, int B)
{
    const int lane = threadIdx.x & 31;
    const int wid  = (int)((blockIdx.x * blockDim.x + threadIdx.x) >> 5);
    const int R0   = wid * 16;              // first batch row of this wave's tile
    if (R0 >= B) return;                    // uniform per wave (grid sized exactly)

    const int h = lane >> 4;                // which half of the wave
    const int m = lane & 15;                // A-row / B-column / D-column index

    // ---- A matrices (16x32 f16 layout, §7.12.2):
    //   lanes 0-15 : slots 0..7 = K 0..7,  slots 8..15 = K16..23 (pad 0)
    //   lanes 16-31: slots 0..7 = K 8..15, slots 8..15 = K24..31 (pad 0)
    // Only K=0..9 are live (N_WIRES), the rest stay zero.
    v16h aHi = {}, aLo = {}, aNg = {};
    const int nj = (h == 0) ? 8 : 2;        // live slots per lane
    for (int s = 0; s < 8; ++s) {
        float lc = 0.0f, ng = 0.0f;
        if (s < nj) {
            const int j = (h == 0) ? s : (8 + s);
            const float c = cosf(x[(R0 + m) * N_WIRES + j]) * cosf(params[j]);
            ng = (c < 0.0f) ? 1.0f : 0.0f;
            lc = fmaxf(logf(fabsf(c)), -60.0f);   // clamp: exp(-60) ~ 0, no inf/NaN
        }
        const _Float16 hi = (_Float16)lc;
        const _Float16 lo = (_Float16)(lc - (float)hi);  // compensated residual
        aHi[s] = hi;
        aLo[s] = lo;
        aNg[s] = (_Float16)ng;
        // slots 8..15 remain zero (K >= 16 padding)
    }

    // ---- B matrix (32x16 f16): lanes 0-15 hold K=0..7,16..23, lanes 16-31 K=8..15,24..31.
    // Prefix mask: B[K,N] = (K <= N && K < N_WIRES) ? 1 : 0  (exact in f16).
    v16h bM = {};
    for (int s = 0; s < 16; ++s) {
        const int K = (s < 8) ? (8 * h + s) : (16 + 8 * h + (s - 8));
        bM[s] = (_Float16)((K <= m && K < N_WIRES) ? 1.0f : 0.0f);
    }

    // ---- Matrix-pipe reductions (EXEC is all-ones here).
    v8f S = {};   // prefix sums of log-magnitudes (fp32 accumulate)
    S = __builtin_amdgcn_wmma_f32_16x16x32_f16(false, aHi, false, bM, (short)0, S, false, false);
    S = __builtin_amdgcn_wmma_f32_16x16x32_f16(false, aLo, false, bM, (short)0, S, false, false);
    v8f P = {};   // prefix counts of negative factors (exact integers)
    P = __builtin_amdgcn_wmma_f32_16x16x32_f16(false, aNg, false, bM, (short)0, P, false, false);

    // ---- Epilogue: D layout = VGPR v -> row (v + 8h), column m. Store k<10 only.
    if (m < N_WIRES) {
        for (int v = 0; v < 8; ++v) {
            const int row   = R0 + v + 8 * h;
            const int par   = ((int)(P[v] + 0.5f)) & 1;
            float val       = expf(S[v]);
            out[row * N_WIRES + m] = par ? -val : val;
        }
    }
}

extern "C" void kernel_launch(void* const* d_in, const int* in_sizes, int n_in,
                              void* d_out, int out_size, void* d_ws, size_t ws_size,
                              hipStream_t stream) {
    const float* x      = (const float*)d_in[0];   // (B, 10) float32
    const float* params = (const float*)d_in[1];   // (10,)   float32
    float*       out    = (float*)d_out;           // (B, 10) float32

    const int B     = in_sizes[0] / N_WIRES;       // 16384
    const int waves = (B + 15) / 16;               // one wave per 16 rows
    const int tpb   = 256;                         // 8 waves / block
    const int blocks = (waves * 32 + tpb - 1) / tpb;

    qsim_prefix_wmma<<<blocks, tpb, 0, stream>>>(x, params, out, B);
}
